// Model_7876970021390
// MI455X (gfx1250) — matrix operand order repaired and checked by reading, not compiled, not run on hardware
//
#include <hip/hip_runtime.h>
#include <hip/hip_bf16.h>

// ---------------- problem constants (match reference) ----------------
#define NV_    4096
#define B_     128
#define NC_    32
#define NS_    128
#define NSITE_ 16384
#define EV_    16384
#define ES_    131072
#define FDIM_  82
#define SDIM_  32
#define H_     128
#define HS_    128
#define HP_    128
#define HI_    64
#define KH_    8
#define P_     524288

typedef __attribute__((ext_vector_type(16))) unsigned short v16us;
typedef __attribute__((ext_vector_type(8)))  unsigned short v8us;
typedef __attribute__((ext_vector_type(16))) __bf16         v16bf;
typedef __attribute__((ext_vector_type(8)))  float          v8f;

__device__ __forceinline__ unsigned short f2bf(float f) {
  // round-to-nearest-even fp32 -> bf16
  unsigned u = __float_as_uint(f);
  u += 0x7FFFu + ((u >> 16) & 1u);
  return (unsigned short)(u >> 16);
}

__device__ __forceinline__ v8f wmma_bf16(v16us a, v16us b, v8f c) {
  return __builtin_amdgcn_wmma_f32_16x16x32_bf16(
      false, __builtin_bit_cast(v16bf, a),
      false, __builtin_bit_cast(v16bf, b),
      (short)0, c, false, false);
}

__device__ __forceinline__ float act_apply(float x, int act) {
  if (act == 1) return x > 0.f ? x : 0.1f * x;        // leaky(0.1)
  if (act == 2) return tanhf(x);
  if (act == 3) return 1.f / (1.f + expf(-x));        // sigmoid
  return x;
}

// ---------------------------------------------------------------------
// Generic WMMA GEMM:  C = act(A @ B + bias)
//   A: [M,K] fp32, row stride lda
//   B: [K,N] (transB=0) or [N,K] (transB=1), row stride ldb
//   C: fp32, element (m,n) at C[m*ldc + n*csc]
//   batching: z = blockIdx.z; zi=z/inner, zb=z%inner; ptr += zi*s?1 + zb*s?2
// Block: 128 threads = 4 wave32, block tile 64x64, wave tile 32x32 (2x2 WMMA),
// BK=64 (two v_wmma_f32_16x16x32_bf16 K-steps per LDS stage).
// LDS tiles are bf16, n/m-major with 72-short rows (144B, 16B aligned) so each
// WMMA fragment = two contiguous ds_load_b128.
// ---------------------------------------------------------------------
#define ROW_ 72
__global__ __launch_bounds__(128) void k_gemm(
    const float* __restrict__ A, int lda, long long sA1, long long sA2,
    const float* __restrict__ Bm, int ldb, int transB, long long sB1, long long sB2,
    const float* __restrict__ bias, long long sbias,
    float* __restrict__ C, int ldc, int csc, long long sC1, long long sC2,
    int M, int N, int Kd, int act, int inner) {
  __shared__ unsigned short As[64][ROW_];   // [m][k] bf16
  __shared__ unsigned short Bs[64][ROW_];   // [n][k] bf16

  int zi = blockIdx.z / inner, zb = blockIdx.z % inner;
  A  += zi * sA1 + zb * sA2;
  Bm += zi * sB1 + zb * sB2;
  if (bias) bias += zi * sbias;
  C  += zi * sC1 + zb * sC2;

  int n0 = blockIdx.x * 64, m0 = blockIdx.y * 64;
  int t = threadIdx.x;
  int lane = t & 31, wid = t >> 5;
  int wm = (wid >> 1) * 32, wn = (wid & 1) * 32;
  int ml = lane & 15, half = lane >> 4;

  v8f z8 = {0.f, 0.f, 0.f, 0.f, 0.f, 0.f, 0.f, 0.f};
  v8f accs[2][2] = {{z8, z8}, {z8, z8}};

  const bool fa_ok = ((lda & 3) == 0) && (m0 + 64 <= M);
  const bool fb_ok = ((ldb & 3) == 0) && (n0 + 64 <= N);

  for (int k0 = 0; k0 < Kd; k0 += 64) {
    bool kfull = (k0 + 64 <= Kd);
    // ---- stage A tile 64x64 (fp32 -> bf16), rows m-major ----
    if (fa_ok && kfull) {
#pragma unroll
      for (int i = 0; i < 8; ++i) {
        int l = (t + i * 128) * 4;            // 4096 floats
        int m = l >> 6, kk = l & 63;
        float4 v = *(const float4*)(A + (long long)(m0 + m) * lda + (k0 + kk));
        As[m][kk + 0] = f2bf(v.x); As[m][kk + 1] = f2bf(v.y);
        As[m][kk + 2] = f2bf(v.z); As[m][kk + 3] = f2bf(v.w);
      }
    } else {
      for (int idx = t; idx < 64 * 64; idx += 128) {
        int m = idx >> 6, kk = idx & 63;
        int gm = m0 + m, gk = k0 + kk;
        int cm = gm < M ? gm : M - 1;
        int ck = gk < Kd ? gk : Kd - 1;
        float v = A[(long long)cm * lda + ck];           // always in-bounds
        As[m][kk] = f2bf((gm < M && gk < Kd) ? v : 0.f); // branchless select
      }
    }
    // ---- stage B tile as Bs[n][k] ----
    if (transB) {
      if (fb_ok && kfull) {
#pragma unroll
        for (int i = 0; i < 8; ++i) {
          int l = (t + i * 128) * 4;
          int n = l >> 6, kk = l & 63;
          float4 v = *(const float4*)(Bm + (long long)(n0 + n) * ldb + (k0 + kk));
          Bs[n][kk + 0] = f2bf(v.x); Bs[n][kk + 1] = f2bf(v.y);
          Bs[n][kk + 2] = f2bf(v.z); Bs[n][kk + 3] = f2bf(v.w);
        }
      } else {
        for (int idx = t; idx < 64 * 64; idx += 128) {
          int n = idx >> 6, kk = idx & 63;
          int gn = n0 + n, gk = k0 + kk;
          int cn = gn < N ? gn : N - 1;
          int ck = gk < Kd ? gk : Kd - 1;
          float v = Bm[(long long)cn * ldb + ck];
          Bs[n][kk] = f2bf((gn < N && gk < Kd) ? v : 0.f);
        }
      }
    } else {
      if (fb_ok && kfull) {
#pragma unroll
        for (int i = 0; i < 8; ++i) {
          int l = (t + i * 128) * 4;            // k-major rows of 64 n
          int kk = l >> 6, n = l & 63;
          float4 v = *(const float4*)(Bm + (long long)(k0 + kk) * ldb + (n0 + n));
          Bs[n + 0][kk] = f2bf(v.x); Bs[n + 1][kk] = f2bf(v.y);
          Bs[n + 2][kk] = f2bf(v.z); Bs[n + 3][kk] = f2bf(v.w);
        }
      } else {
        for (int idx = t; idx < 64 * 64; idx += 128) {
          int kk = idx >> 6, n = idx & 63;
          int gk = k0 + kk, gn = n0 + n;
          int ck = gk < Kd ? gk : Kd - 1;
          int cn = gn < N ? gn : N - 1;
          float v = Bm[(long long)ck * ldb + cn];
          Bs[n][kk] = f2bf((gk < Kd && gn < N) ? v : 0.f);
        }
      }
    }
    __syncthreads();

    // ---- two WMMA K-steps of 32 ----
#pragma unroll
    for (int ks = 0; ks < 2; ++ks) {
      int kb = ks * 32;
      // fragment = concat of two contiguous 8-short runs -> 2x ds_load_b128
      v8us a0l = *(const v8us*)&As[wm + ml][kb + 8 * half];
      v8us a0h = *(const v8us*)&As[wm + ml][kb + 16 + 8 * half];
      v8us a1l = *(const v8us*)&As[wm + 16 + ml][kb + 8 * half];
      v8us a1h = *(const v8us*)&As[wm + 16 + ml][kb + 16 + 8 * half];
      v8us b0l = *(const v8us*)&Bs[wn + ml][kb + 8 * half];
      v8us b0h = *(const v8us*)&Bs[wn + ml][kb + 16 + 8 * half];
      v8us b1l = *(const v8us*)&Bs[wn + 16 + ml][kb + 8 * half];
      v8us b1h = *(const v8us*)&Bs[wn + 16 + ml][kb + 16 + 8 * half];
      v16us afr[2], bfr[2];
      afr[0] = __builtin_shufflevector(a0l, a0h, 0,1,2,3,4,5,6,7,8,9,10,11,12,13,14,15);
      afr[1] = __builtin_shufflevector(a1l, a1h, 0,1,2,3,4,5,6,7,8,9,10,11,12,13,14,15);
      bfr[0] = __builtin_shufflevector(b0l, b0h, 0,1,2,3,4,5,6,7,8,9,10,11,12,13,14,15);
      bfr[1] = __builtin_shufflevector(b1l, b1h, 0,1,2,3,4,5,6,7,8,9,10,11,12,13,14,15);
#pragma unroll
      for (int i = 0; i < 2; ++i)
#pragma unroll
        for (int j = 0; j < 2; ++j)
          accs[i][j] = wmma_bf16(afr[i], bfr[j], accs[i][j]);
    }
    __syncthreads();
  }

  // epilogue: D layout lane L vgpr v -> m = v + 8*(L>>4), n = L&15
#pragma unroll
  for (int i = 0; i < 2; ++i)
#pragma unroll
    for (int j = 0; j < 2; ++j)
#pragma unroll
      for (int v = 0; v < 8; ++v) {
        int row = m0 + wm + 16 * i + v + 8 * half;
        int col = n0 + wn + 16 * j + ml;
        if (row < M && col < N) {
          float x = accs[i][j][v];
          if (bias) x += bias[col];
          C[(long long)row * ldc + (long long)col * csc] = act_apply(x, act);
        }
      }
}

// ---------------- helper kernels (gather/scatter/elementwise) ----------------
__global__ void k_zero(float* __restrict__ p, long long n) {
  long long id = (long long)blockIdx.x * blockDim.x + threadIdx.x;
  if (id < n) p[id] = 0.f;
}

__global__ void k_add(const float* __restrict__ a, const float* __restrict__ b,
                      float* __restrict__ o, long long n) {
  long long id = (long long)blockIdx.x * blockDim.x + threadIdx.x;
  if (id < n) o[id] = a[id] + b[id];
}

__global__ void k_copy(const float* __restrict__ src, int lds_,
                       float* __restrict__ dst, int ldd, int M, int cols) {
  long long id = (long long)blockIdx.x * blockDim.x + threadIdx.x;
  if (id >= (long long)M * cols) return;
  int r = (int)(id / cols), c = (int)(id % cols);
  dst[(long long)r * ldd + c] = src[(long long)r * lds_ + c];
}

__global__ void k_zero_cols(float* __restrict__ dst, int ldd, int M, int cols) {
  long long id = (long long)blockIdx.x * blockDim.x + threadIdx.x;
  if (id >= (long long)M * cols) return;
  int r = (int)(id / cols), c = (int)(id % cols);
  dst[(long long)r * ldd + c] = 0.f;
}

// edge aggregation: out[dst[e]] += rows[src[e]]  (random indices -> f32 atomics)
__global__ void k_scatter(const float* __restrict__ rows, int ldr,
                          const int* __restrict__ src, const int* __restrict__ dst,
                          float* __restrict__ outp, int ldo, int dim, long long total) {
  long long id = (long long)blockIdx.x * blockDim.x + threadIdx.x;
  if (id >= total) return;
  int e = (int)(id / dim), c = (int)(id % dim);
  int s = src[e], d = dst[e];
  atomicAdd(&outp[(long long)d * ldo + c], rows[(long long)s * ldr + c]);
}

// super node init: contiguous 32-row segment sum
__global__ void k_seg32(const float* __restrict__ v, float* __restrict__ s) {
  int b = blockIdx.x, e = threadIdx.x;
  float acc = 0.f;
  for (int c = 0; c < 32; ++c) acc += v[((long long)b * 32 + c) * H_ + e];
  s[b * H_ + e] = acc;
}

// score[n,k] = sum_e am[k,n,e]*asup[k,b(n),e]*Wb[k,e]
__global__ void k_score(const float* __restrict__ am, const float* __restrict__ asup,
                        const float* __restrict__ Wb, float* __restrict__ score) {
  int id = blockIdx.x * blockDim.x + threadIdx.x;
  if (id >= NV_ * KH_) return;
  int k = id & 7, n = id >> 3, b = n >> 5;
  const float* amp = am + ((long long)k * NV_ + n) * H_;
  const float* asp = asup + ((long long)k * B_ + b) * H_;
  const float* wbp = Wb + k * H_;
  float acc = 0.f;
  for (int e = 0; e < H_; ++e) acc += amp[e] * asp[e] * wbp[e];
  score[n * KH_ + k] = acc;
}

// per-(graph,head) softmax over 32 contiguous vertices
__global__ void k_smax(const float* __restrict__ score, float* __restrict__ attn) {
  int id = blockIdx.x * blockDim.x + threadIdx.x;
  if (id >= B_ * KH_) return;
  int k = id & 7, b = id >> 3;
  float mx = -1e30f;
  for (int c = 0; c < 32; ++c) mx = fmaxf(mx, score[(b * 32 + c) * KH_ + k]);
  float sum = 0.f;
  for (int c = 0; c < 32; ++c) sum += expf(score[(b * 32 + c) * KH_ + k] - mx);
  float inv = 1.f / sum;
  for (int c = 0; c < 32; ++c)
    attn[(b * 32 + c) * KH_ + k] = expf(score[(b * 32 + c) * KH_ + k] - mx) * inv;
}

// m2s[b,k,e] = sum_c attn[b*32+c,k]*msg[k,b*32+c,e]
__global__ void k_m2s(const float* __restrict__ attn, const float* __restrict__ msg,
                      float* __restrict__ m2s) {
  int id = blockIdx.x * blockDim.x + threadIdx.x;
  if (id >= B_ * KH_ * H_) return;
  int e = id & 127, k = (id >> 7) & 7, b = id >> 10;
  float acc = 0.f;
  for (int c = 0; c < 32; ++c) {
    int n = b * 32 + c;
    acc += attn[n * KH_ + k] * msg[((long long)k * NV_ + n) * H_ + e];
  }
  m2s[id] = acc;  // id == b*K*H + k*H + e  (matches reshape(B, K*H))
}

// warp gate: z=sig(g1[m]+g2[m/div]); x=(1-z)*a[m]+z*bb[m/div]
__global__ void k_gate(const float* __restrict__ g1, const float* __restrict__ g2,
                       const float* __restrict__ a, const float* __restrict__ bb,
                       float* __restrict__ x, int M, int Hh, int div) {
  long long id = (long long)blockIdx.x * blockDim.x + threadIdx.x;
  if (id >= (long long)M * Hh) return;
  int m = (int)(id / Hh), e = (int)(id % Hh);
  int mb = m / div;
  float z = 1.f / (1.f + expf(-(g1[id] + g2[(long long)mb * Hh + e])));
  x[id] = (1.f - z) * a[id] + z * bb[(long long)mb * Hh + e];
}

// GRU combine (in-place on h): gi,gh are [M,3H]
__global__ void k_gru(const float* __restrict__ gi, const float* __restrict__ gh,
                      float* __restrict__ h, int M, int Hh) {
  long long id = (long long)blockIdx.x * blockDim.x + threadIdx.x;
  if (id >= (long long)M * Hh) return;
  int m = (int)(id / Hh), e = (int)(id % Hh);
  long long base = (long long)m * 3 * Hh;
  float r = 1.f / (1.f + expf(-(gi[base + e] + gh[base + e])));
  float z = 1.f / (1.f + expf(-(gi[base + Hh + e] + gh[base + Hh + e])));
  float n = tanhf(gi[base + 2 * Hh + e] + r * gh[base + 2 * Hh + e]);
  h[id] = (1.f - z) * n + z * h[id];
}

// ---------------------------------------------------------------------
extern "C" void kernel_launch(void* const* d_in, const int* in_sizes, int n_in,
                              void* d_out, int out_size, void* d_ws, size_t ws_size,
                              hipStream_t stream) {
  (void)in_sizes; (void)n_in; (void)out_size; (void)ws_size;
  const float* x_vert  = (const float*)d_in[0];
  const int*   vsrc    = (const int*)d_in[1];
  const int*   vdst    = (const int*)d_in[2];
  /* d_in[3] vert_batch: contiguous repeat(arange(B),32) -> use n>>5 */
  const float* W_emb   = (const float*)d_in[4];
  const float* b_emb   = (const float*)d_in[5];
  const float* Wa_main = (const float*)d_in[6];
  const float* Wa_sup  = (const float*)d_in[7];
  const float* Wv      = (const float*)d_in[8];
  const float* Wb      = (const float*)d_in[9];
  const float* Wm2s    = (const float*)d_in[10];
  const float* Ws2m    = (const float*)d_in[11];
  const float* Wsup    = (const float*)d_in[12];
  const float* Wself   = (const float*)d_in[13];
  const float* Wzm1    = (const float*)d_in[14];
  const float* Wzm2    = (const float*)d_in[15];
  const float* Wzs1    = (const float*)d_in[16];
  const float* Wzs2    = (const float*)d_in[17];
  const float* Wih_m   = (const float*)d_in[18];
  const float* Whh_m   = (const float*)d_in[19];
  const float* bih_m   = (const float*)d_in[20];
  const float* bhh_m   = (const float*)d_in[21];
  const float* Wih_s   = (const float*)d_in[22];
  const float* Whh_s   = (const float*)d_in[23];
  const float* bih_s   = (const float*)d_in[24];
  const float* bhh_s   = (const float*)d_in[25];
  const float* x_site  = (const float*)d_in[26];
  const int*   ssrc    = (const int*)d_in[27];
  const int*   sdst    = (const int*)d_in[28];
  const float* W0      = (const float*)d_in[29];
  const float* b0      = (const float*)d_in[30];
  const float* Wconv   = (const float*)d_in[31];
  const float* bconv   = (const float*)d_in[32];
  const float* Wout    = (const float*)d_in[33];
  const float* bout    = (const float*)d_in[34];
  const float* Wis     = (const float*)d_in[35];
  const float* bis     = (const float*)d_in[36];
  const float* Wic     = (const float*)d_in[37];
  const float* bic     = (const float*)d_in[38];
  float* out = (float*)d_out;

  // ---- workspace: persistent buffers + phase-reused arena ----
  float* Wsp = (float*)d_ws;
  size_t off = 0;
  auto alloc = [&](size_t n) { float* p = Wsp + off; off += n; return p; };
  float* vert     = alloc((size_t)NV_ * H_);      // persists to interaction phase
  float* site_out = alloc((size_t)NSITE_ * HP_);  // persists to interaction phase
  float* arena = Wsp + off;

  auto gemm = [&](const float* A, int lda, long long sA1, long long sA2,
                  const float* Bm, int ldb, int tB, long long sB1, long long sB2,
                  const float* bias, long long sbias,
                  float* Cc, int ldc, int csc, long long sC1, long long sC2,
                  int M, int N, int Kd, int act, int nz, int inner) {
    dim3 g((unsigned)((N + 63) / 64), (unsigned)((M + 63) / 64), (unsigned)nz);
    k_gemm<<<g, 128, 0, stream>>>(A, lda, sA1, sA2, Bm, ldb, tB, sB1, sB2,
                                  bias, sbias, Cc, ldc, csc, sC1, sC2,
                                  M, N, Kd, act, inner);
  };
  auto EL = [](long long n) { return dim3((unsigned)((n + 255) / 256)); };

  // ================= Phase 1: GWM compound GNN =================
  size_t co = 0;
  auto ca = [&](size_t n) { float* p = arena + co; co += n; return p; };
  float* supe  = ca((size_t)B_ * H_);
  float* nei   = ca((size_t)NV_ * H_);
  float* tmp   = ca((size_t)NV_ * H_);
  float* mself = ca((size_t)NV_ * H_);
  float* am    = ca((size_t)KH_ * NV_ * H_);
  float* asup  = ca((size_t)KH_ * B_ * H_);
  float* msg   = ca((size_t)KH_ * NV_ * H_);
  float* score = ca((size_t)NV_ * KH_);
  float* attn  = ca((size_t)NV_ * KH_);
  float* m2s   = ca((size_t)B_ * KH_ * H_);
  float* mts   = ca((size_t)B_ * H_);
  float* s2m   = ca((size_t)B_ * H_);
  float* g1    = ca((size_t)NV_ * H_);
  float* g2    = ca((size_t)B_ * H_);
  float* xm    = ca((size_t)NV_ * H_);
  float* gi    = ca((size_t)NV_ * 3 * H_);
  float* gh    = ca((size_t)NV_ * 3 * H_);
  float* ssn   = ca((size_t)B_ * H_);
  float* g1s   = ca((size_t)B_ * H_);
  float* g2s   = ca((size_t)B_ * H_);
  float* xs    = ca((size_t)B_ * H_);
  float* gis   = ca((size_t)B_ * 3 * H_);
  float* ghs   = ca((size_t)B_ * 3 * H_);

  // embedding: vert = leaky(x_vert @ W_emb + b_emb)   (K=82, lda=82 -> guarded path)
  gemm(x_vert, FDIM_, 0, 0, W_emb, H_, 0, 0, 0, b_emb, 0,
       vert, H_, 1, 0, 0, NV_, H_, FDIM_, 1, 1, 1);
  k_seg32<<<B_, H_, 0, stream>>>(vert, supe);

  for (int d = 0; d < 3; ++d) {
    const float* Wself_d = Wself + (size_t)d * H_ * H_;
    const float* Wam_d   = Wa_main + (size_t)d * KH_ * H_ * H_;
    const float* Was_d   = Wa_sup + (size_t)d * KH_ * H_ * H_;
    const float* Wv_d    = Wv + (size_t)d * KH_ * H_ * H_;
    const float* Wb_d    = Wb + (size_t)d * KH_ * H_;
    const float* Wm2s_d  = Wm2s + (size_t)d * KH_ * H_ * H_;
    const float* Ws2m_d  = Ws2m + (size_t)d * H_ * H_;
    const float* Wsup_d  = Wsup + (size_t)d * H_ * H_;
    const float* Wzm1_d  = Wzm1 + (size_t)d * H_ * H_;
    const float* Wzm2_d  = Wzm2 + (size_t)d * H_ * H_;
    const float* Wzs1_d  = Wzs1 + (size_t)d * H_ * H_;
    const float* Wzs2_d  = Wzs2 + (size_t)d * H_ * H_;

    // local message passing + self transform
    k_zero<<<EL((long long)NV_ * H_), 256, 0, stream>>>(nei, (long long)NV_ * H_);
    k_scatter<<<EL((long long)EV_ * H_), 256, 0, stream>>>(
        vert, H_, vsrc, vdst, nei, H_, H_, (long long)EV_ * H_);
    k_add<<<EL((long long)NV_ * H_), 256, 0, stream>>>(vert, nei, tmp, (long long)NV_ * H_);
    gemm(tmp, H_, 0, 0, Wself_d, H_, 0, 0, 0, nullptr, 0,
         mself, H_, 1, 0, 0, NV_, H_, H_, 1, 1, 1);

    // multi-head attention (head-batched WMMA GEMMs)
    gemm(vert, H_, 0, 0, Wam_d, H_, 0, (long long)H_ * H_, 0, nullptr, 0,
         am, H_, 1, (long long)NV_ * H_, 0, NV_, H_, H_, 2, KH_, 1);
    gemm(supe, H_, 0, 0, Was_d, H_, 0, (long long)H_ * H_, 0, nullptr, 0,
         asup, H_, 1, (long long)B_ * H_, 0, B_, H_, H_, 2, KH_, 1);
    k_score<<<EL(NV_ * KH_), 256, 0, stream>>>(am, asup, Wb_d, score);
    k_smax<<<EL(B_ * KH_), 256, 0, stream>>>(score, attn);
    gemm(vert, H_, 0, 0, Wv_d, H_, 0, (long long)H_ * H_, 0, nullptr, 0,
         msg, H_, 1, (long long)NV_ * H_, 0, NV_, H_, H_, 0, KH_, 1);
    k_m2s<<<EL(B_ * KH_ * H_), 256, 0, stream>>>(attn, msg, m2s);
    gemm(m2s, KH_ * H_, 0, 0, Wm2s_d, H_, 0, 0, 0, nullptr, 0,
         mts, H_, 1, 0, 0, B_, H_, KH_ * H_, 2, 1, 1);
    gemm(supe, H_, 0, 0, Ws2m_d, H_, 0, 0, 0, nullptr, 0,
         s2m, H_, 1, 0, 0, B_, H_, H_, 2, 1, 1);

    // main-side warp gate + GRU
    gemm(mself, H_, 0, 0, Wzm1_d, H_, 0, 0, 0, nullptr, 0,
         g1, H_, 1, 0, 0, NV_, H_, H_, 0, 1, 1);
    gemm(s2m, H_, 0, 0, Wzm2_d, H_, 0, 0, 0, nullptr, 0,
         g2, H_, 1, 0, 0, B_, H_, H_, 0, 1, 1);
    k_gate<<<EL((long long)NV_ * H_), 256, 0, stream>>>(g1, g2, mself, s2m, xm, NV_, H_, 32);
    gemm(xm, H_, 0, 0, Wih_m, 3 * H_, 0, 0, 0, bih_m, 0,
         gi, 3 * H_, 1, 0, 0, NV_, 3 * H_, H_, 0, 1, 1);
    gemm(vert, H_, 0, 0, Whh_m, 3 * H_, 0, 0, 0, bhh_m, 0,
         gh, 3 * H_, 1, 0, 0, NV_, 3 * H_, H_, 0, 1, 1);

    // super-side warp gate + GRU (uses old supe; launched before GRU writes)
    gemm(supe, H_, 0, 0, Wsup_d, H_, 0, 0, 0, nullptr, 0,
         ssn, H_, 1, 0, 0, B_, H_, H_, 2, 1, 1);
    gemm(ssn, H_, 0, 0, Wzs1_d, H_, 0, 0, 0, nullptr, 0,
         g1s, H_, 1, 0, 0, B_, H_, H_, 0, 1, 1);
    gemm(mts, H_, 0, 0, Wzs2_d, H_, 0, 0, 0, nullptr, 0,
         g2s, H_, 1, 0, 0, B_, H_, H_, 0, 1, 1);
    k_gate<<<EL((long long)B_ * H_), 256, 0, stream>>>(g1s, g2s, ssn, mts, xs, B_, H_, 1);
    gemm(xs, H_, 0, 0, Wih_s, 3 * H_, 0, 0, 0, bih_s, 0,
         gis, 3 * H_, 1, 0, 0, B_, 3 * H_, H_, 0, 1, 1);
    gemm(supe, H_, 0, 0, Whh_s, 3 * H_, 0, 0, 0, bhh_s, 0,
         ghs, 3 * H_, 1, 0, 0, B_, 3 * H_, H_, 0, 1, 1);

    k_gru<<<EL((long long)NV_ * H_), 256, 0, stream>>>(gi, gh, vert, NV_, H_);
    k_gru<<<EL((long long)B_ * H_), 256, 0, stream>>>(gis, ghs, supe, B_, H_);
  }

  // ================= Phase 2: AtomConv protein GNN =================
  co = 0;  // reuse arena (vert/site_out live outside it)
  float* feats = ca((size_t)NSITE_ * 416);   // [x(32) | h1(128) | h2(128) | h3(128)]
  float* cat   = ca((size_t)NSITE_ * 256);   // [h | nei] staging

  k_copy<<<EL((long long)NSITE_ * 32), 256, 0, stream>>>(x_site, SDIM_, feats, 416, NSITE_, 32);

  // layer 0: K = 64
  k_copy<<<EL((long long)NSITE_ * 32), 256, 0, stream>>>(x_site, SDIM_, cat, 64, NSITE_, 32);
  k_zero_cols<<<EL((long long)NSITE_ * 32), 256, 0, stream>>>(cat + 32, 64, NSITE_, 32);
  k_scatter<<<EL((long long)ES_ * 32), 256, 0, stream>>>(
      x_site, SDIM_, ssrc, sdst, cat + 32, 64, 32, (long long)ES_ * 32);
  gemm(cat, 64, 0, 0, W0, HS_, 0, 0, 0, b0, 0,
       feats + 32, 416, 1, 0, 0, NSITE_, HS_, 64, 1, 1, 1);

  // layers 1,2: K = 256
  for (int i = 0; i < 2; ++i) {
    const float* hsrc = feats + 32 + (size_t)i * 128;
    float* hdst = feats + 160 + (size_t)i * 128;
    k_copy<<<EL((long long)NSITE_ * 128), 256, 0, stream>>>(hsrc, 416, cat, 256, NSITE_, 128);
    k_zero_cols<<<EL((long long)NSITE_ * 128), 256, 0, stream>>>(cat + 128, 256, NSITE_, 128);
    k_scatter<<<EL((long long)ES_ * 128), 256, 0, stream>>>(
        hsrc, 416, ssrc, sdst, cat + 128, 256, 128, (long long)ES_ * 128);
    gemm(cat, 256, 0, 0, Wconv + (size_t)i * 256 * HS_, HS_, 0, 0, 0,
         bconv + (size_t)i * HS_, 0,
         hdst, 416, 1, 0, 0, NSITE_, HS_, 256, 1, 1, 1);
  }

  // site_out = leaky(feats @ Wout + bout), K = 416 (6*64 + 32 tail)
  gemm(feats, 416, 0, 0, Wout, HP_, 0, 0, 0, bout, 0,
       site_out, HP_, 1, 0, 0, NSITE_, HP_, 416, 1, 1, 1);

  // ================= Phase 3: 7-head interaction =================
  co = 0;  // reuse arena again (feats/cat dead)
  float* pa = ca((size_t)7 * NSITE_ * HI_);
  float* pg = ca((size_t)7 * NV_ * HI_);

  gemm(site_out, HP_, 0, 0, Wis, HI_, 0, (long long)HP_ * HI_, 0, bis, HI_,
       pa, HI_, 1, (long long)NSITE_ * HI_, 0, NSITE_, HI_, HP_, 1, 7, 1);
  gemm(vert, H_, 0, 0, Wic, HI_, 0, (long long)H_ * HI_, 0, bic, HI_,
       pg, HI_, 1, (long long)NV_ * HI_, 0, NV_, HI_, H_, 1, 7, 1);

  // out[((b*128+s)*32+c)*7+i] = pa[i][b*128+s] . pg[i][b*32+c]
  // batched WMMA GEMM: z = i*128 + b ; A=[128x64], B=[32x64]^T, C strided into [P,7]
  gemm(pa, HI_, (long long)NSITE_ * HI_, (long long)NS_ * HI_,
       pg, HI_, 1, (long long)NV_ * HI_, (long long)NC_ * HI_,
       nullptr, 0,
       out, NC_ * 7, 7, 1, (long long)NS_ * NC_ * 7,
       NS_, NC_, HI_, 0, 7 * B_, B_);
}
